// CausalAttention_44255343018151
// MI455X (gfx1250) — matrix operand order repaired and checked by reading, not compiled, not run on hardware
//
#include <hip/hip_runtime.h>

// ---------------------------------------------------------------------------
// Causal attention forward for MI455X (gfx1250, wave32, WMMA 16x16x32 f16)
//   out = [att_output (B*S*D f32) | att_weights (B*S*S f32)]
// Register-blocked WMMA GEMMs: 32x64 tiles (proj / AV), 32x32 tiles (scores).
// ---------------------------------------------------------------------------

#define BB   4
#define SS   2048
#define DD   1024
#define BSR  (BB * SS)          // 8192 rows of x / q / k / v

typedef __attribute__((ext_vector_type(16))) _Float16 v16h;
typedef __attribute__((ext_vector_type(8)))  float    v8f;

union Frag16 { uint4 q[2]; v16h v; };

// A-matrix fragment (16x32 f16), src row-major, base -> (row0, k0), stride ld.
// lanes 0-15 = rows M=0..15; h=lane>>4 selects K-halves:
//   elems 0..7  -> K = h*8 + j,   elems 8..15 -> K = 16 + h*8 + j
__device__ __forceinline__ v16h load_frag_a(const _Float16* base, int ld, int lane) {
  const int m = lane & 15, h = lane >> 4;
  const _Float16* p = base + (size_t)m * ld + h * 8;
  Frag16 u;
  u.q[0] = *(const uint4*)(p);
  u.q[1] = *(const uint4*)(p + 16);
  return u.v;
}

// B-matrix fragment (32x16 f16) from N-major storage: element (k,n) at base[n*ld+k].
// h=lane>>4: elems 0..7 -> K = h*8 + j, elems 8..15 -> K = 16 + h*8 + j (matches A).
__device__ __forceinline__ v16h load_frag_b(const _Float16* base, int ld, int lane) {
  const int n = lane & 15, h = lane >> 4;
  const _Float16* p = base + (size_t)n * ld + h * 8;
  Frag16 u;
  u.q[0] = *(const uint4*)(p);
  u.q[1] = *(const uint4*)(p + 16);
  return u.v;
}

#define WMMA_F16(a, b, c) \
  __builtin_amdgcn_wmma_f32_16x16x32_f16(false, (a), false, (b), (short)0, (c), false, false)

// ------------------------- kernel 1a: x -> f16 ------------------------------
__global__ __launch_bounds__(256) void cvt_x_kernel(const float* __restrict__ x,
                                                    _Float16* __restrict__ xh) {
  size_t i = (size_t)blockIdx.x * blockDim.x + threadIdx.x;
  xh[i] = (_Float16)x[i];
}

// ---------------- kernel 1b: W[k][n] -> Wt[n][k] (f16, N-major) -------------
__global__ __launch_bounds__(256) void cvt_w_kernel(const float* __restrict__ Wq,
                                                    const float* __restrict__ Wk,
                                                    const float* __restrict__ Wv,
                                                    _Float16* __restrict__ wqT,
                                                    _Float16* __restrict__ wkT,
                                                    _Float16* __restrict__ wvT) {
  size_t i = (size_t)blockIdx.x * blockDim.x + threadIdx.x;   // i = k*DD + n
  int k = (int)(i / DD), n = (int)(i % DD);
  size_t o = (size_t)n * DD + k;
  wqT[o] = (_Float16)Wq[i];
  wkT[o] = (_Float16)Wk[i];
  wvT[o] = (_Float16)Wv[i];
}

// ------------------- kernel 2: q/k/v = x @ W  (WMMA, 32x64/wave) ------------
// blockIdx.y = which (0:q 1:k 2:v); tile -> {mb (256), nb (16)}
__global__ __launch_bounds__(128) void qkv_gemm_kernel(const _Float16* __restrict__ xh,
                                                       const _Float16* __restrict__ wqT,
                                                       const _Float16* __restrict__ wkT,
                                                       const _Float16* __restrict__ wvT,
                                                       _Float16* __restrict__ qh,
                                                       _Float16* __restrict__ kh,
                                                       _Float16* __restrict__ vT) {
  const int lane  = threadIdx.x & 31;
  const int which = blockIdx.y;
  const int tile  = blockIdx.x * 4 + (threadIdx.x >> 5);
  const int NB = DD / 64;                           // 16
  const int mb = tile / NB, nb = tile % NB;         // mb: 0..255

  const _Float16* wT = (which == 0) ? wqT : (which == 1) ? wkT : wvT;
  const _Float16* ab = xh + (size_t)(mb * 32) * DD;
  const _Float16* bb = wT + (size_t)(nb * 64) * DD;

  v8f c[2][4] = {};
  for (int k0 = 0; k0 < DD; k0 += 32) {
    v16h a0 = load_frag_a(ab + k0, DD, lane);
    v16h a1 = load_frag_a(ab + (size_t)16 * DD + k0, DD, lane);
    v16h b0 = load_frag_b(bb + k0, DD, lane);
    v16h b1 = load_frag_b(bb + (size_t)16 * DD + k0, DD, lane);
    v16h b2 = load_frag_b(bb + (size_t)32 * DD + k0, DD, lane);
    v16h b3 = load_frag_b(bb + (size_t)48 * DD + k0, DD, lane);
    c[0][0] = WMMA_F16(a0, b0, c[0][0]);
    c[0][1] = WMMA_F16(a0, b1, c[0][1]);
    c[0][2] = WMMA_F16(a0, b2, c[0][2]);
    c[0][3] = WMMA_F16(a0, b3, c[0][3]);
    c[1][0] = WMMA_F16(a1, b0, c[1][0]);
    c[1][1] = WMMA_F16(a1, b1, c[1][1]);
    c[1][2] = WMMA_F16(a1, b2, c[1][2]);
    c[1][3] = WMMA_F16(a1, b3, c[1][3]);
  }

  const int h = lane >> 4, nloc = lane & 15;
#pragma unroll
  for (int i = 0; i < 2; ++i) {
#pragma unroll
    for (int j = 0; j < 4; ++j) {
#pragma unroll
      for (int r = 0; r < 8; ++r) {
        int row = mb * 32 + i * 16 + r + 8 * h;     // [0, BSR)
        int col = nb * 64 + j * 16 + nloc;          // [0, DD)
        _Float16 val = (_Float16)c[i][j][r];
        if (which == 0) {
          qh[(size_t)row * DD + col] = val;
        } else if (which == 1) {
          kh[(size_t)row * DD + col] = val;         // [s][d] == N-major for QK^T
        } else {
          int b = row >> 11, s = row & (SS - 1);    // store V transposed [b][d][s]
          vT[((size_t)b * DD + col) * SS + s] = val;
        }
      }
    }
  }
}

// --------- kernel 3: raw causal scores = (q @ k^T) * scale (32x32/wave) -----
// Upper-triangle blocks (kb > qb) are never read by softmax -> skip entirely.
__global__ __launch_bounds__(128) void scores_kernel(const _Float16* __restrict__ qh,
                                                     const _Float16* __restrict__ kh,
                                                     float* __restrict__ scoresOut) {
  const int lane = threadIdx.x & 31;
  const int b    = blockIdx.y;
  const int tile = blockIdx.x * 4 + (threadIdx.x >> 5);
  const int QB = SS / 32;                           // 64
  const int qb = tile / QB, kb = tile % QB;
  if (kb > qb) return;                              // fully masked: softmax zeros it

  const _Float16* ab = qh + (size_t)(b * SS + qb * 32) * DD;
  const _Float16* bb = kh + (size_t)(b * SS + kb * 32) * DD;

  v8f c[2][2] = {};
  for (int k0 = 0; k0 < DD; k0 += 32) {
    v16h a0 = load_frag_a(ab + k0, DD, lane);
    v16h a1 = load_frag_a(ab + (size_t)16 * DD + k0, DD, lane);
    v16h b0 = load_frag_b(bb + k0, DD, lane);
    v16h b1 = load_frag_b(bb + (size_t)16 * DD + k0, DD, lane);
    c[0][0] = WMMA_F16(a0, b0, c[0][0]);
    c[0][1] = WMMA_F16(a0, b1, c[0][1]);
    c[1][0] = WMMA_F16(a1, b0, c[1][0]);
    c[1][1] = WMMA_F16(a1, b1, c[1][1]);
  }

  float* outb = scoresOut + (size_t)b * SS * SS;
  const float scale = 0.03125f;                     // 1/sqrt(1024)
  const int h = lane >> 4, nloc = lane & 15;
#pragma unroll
  for (int i = 0; i < 2; ++i)
#pragma unroll
    for (int j = 0; j < 2; ++j)
#pragma unroll
      for (int r = 0; r < 8; ++r) {
        int row = qb * 32 + i * 16 + r + 8 * h;
        int col = kb * 32 + j * 16 + nloc;
        outb[(size_t)row * SS + col] = c[i][j][r] * scale;
      }
}

// --------- kernel 4: row softmax (causal length s+1), f32 + f16 copy --------
__global__ __launch_bounds__(256) void softmax_kernel(float* __restrict__ scoresOut,
                                                      _Float16* __restrict__ wh) {
  __shared__ float red[256];
  const int gr = blockIdx.x;                        // 0 .. B*S-1
  const int b = gr / SS, s = gr % SS;
  float*    row  = scoresOut + (size_t)(b * SS + s) * SS;
  _Float16* hrow = wh        + (size_t)(b * SS + s) * SS;
  const int t = threadIdx.x;

  float vals[8], ev[8];
  float mx = -3.4e38f;
#pragma unroll
  for (int i = 0; i < 8; ++i) {
    int col = t + i * 256;
    float v = (col <= s) ? row[col] : -3.4e38f;
    vals[i] = v;
    mx = fmaxf(mx, v);
  }
  red[t] = mx; __syncthreads();
  for (int off = 128; off > 0; off >>= 1) {
    if (t < off) red[t] = fmaxf(red[t], red[t + off]);
    __syncthreads();
  }
  mx = red[0]; __syncthreads();

  float sum = 0.0f;
#pragma unroll
  for (int i = 0; i < 8; ++i) {
    int col = t + i * 256;
    float e = (col <= s) ? __expf(vals[i] - mx) : 0.0f;
    ev[i] = e; sum += e;
  }
  red[t] = sum; __syncthreads();
  for (int off = 128; off > 0; off >>= 1) {
    if (t < off) red[t] += red[t + off];
    __syncthreads();
  }
  const float inv = 1.0f / red[0];

#pragma unroll
  for (int i = 0; i < 8; ++i) {
    int col = t + i * 256;
    float w = (col <= s) ? ev[i] * inv : 0.0f;      // zero beyond causal edge
    row[col]  = w;
    hrow[col] = (_Float16)w;
  }
}

// --------- kernel 5: att_output = weights @ v (WMMA, 32x64/wave) ------------
__global__ __launch_bounds__(128) void av_gemm_kernel(const _Float16* __restrict__ wh,
                                                      const _Float16* __restrict__ vT,
                                                      float* __restrict__ out) {
  const int lane = threadIdx.x & 31;
  const int b    = blockIdx.y;
  const int tile = blockIdx.x * 4 + (threadIdx.x >> 5);
  const int NB = DD / 64;                           // 16
  const int qb = tile / NB, nb = tile % NB;         // qb: 0..63

  const _Float16* ab = wh + (size_t)(b * SS + qb * 32) * SS;
  const _Float16* bb = vT + ((size_t)b * DD + nb * 64) * SS;

  const int ksteps = qb * 32 + 32;                  // causal trim (weights zero past s)
  v8f c[2][4] = {};
  for (int k0 = 0; k0 < ksteps; k0 += 32) {
    v16h a0 = load_frag_a(ab + k0, SS, lane);
    v16h a1 = load_frag_a(ab + (size_t)16 * SS + k0, SS, lane);
    v16h b0 = load_frag_b(bb + k0, SS, lane);
    v16h b1 = load_frag_b(bb + (size_t)16 * SS + k0, SS, lane);
    v16h b2 = load_frag_b(bb + (size_t)32 * SS + k0, SS, lane);
    v16h b3 = load_frag_b(bb + (size_t)48 * SS + k0, SS, lane);
    c[0][0] = WMMA_F16(a0, b0, c[0][0]);
    c[0][1] = WMMA_F16(a0, b1, c[0][1]);
    c[0][2] = WMMA_F16(a0, b2, c[0][2]);
    c[0][3] = WMMA_F16(a0, b3, c[0][3]);
    c[1][0] = WMMA_F16(a1, b0, c[1][0]);
    c[1][1] = WMMA_F16(a1, b1, c[1][1]);
    c[1][2] = WMMA_F16(a1, b2, c[1][2]);
    c[1][3] = WMMA_F16(a1, b3, c[1][3]);
  }

  const int h = lane >> 4, nloc = lane & 15;
#pragma unroll
  for (int i = 0; i < 2; ++i)
#pragma unroll
    for (int j = 0; j < 4; ++j)
#pragma unroll
      for (int r = 0; r < 8; ++r) {
        int row = qb * 32 + i * 16 + r + 8 * h;
        int col = nb * 64 + j * 16 + nloc;
        out[(size_t)(b * SS + row) * DD + col] = c[i][j][r];
      }
}

// ---------------------------------------------------------------------------
extern "C" void kernel_launch(void* const* d_in, const int* in_sizes, int n_in,
                              void* d_out, int out_size, void* d_ws, size_t ws_size,
                              hipStream_t stream) {
  const float* x  = (const float*)d_in[0];
  const float* Wq = (const float*)d_in[1];
  const float* Wk = (const float*)d_in[2];
  const float* Wv = (const float*)d_in[3];

  float* attOut    = (float*)d_out;                       // B*S*D
  float* scoresOut = attOut + (size_t)BSR * DD;           // B*S*S

  // workspace carve-up (f16): ~102 MB total
  _Float16* xh  = (_Float16*)d_ws;                        // BSR*DD
  _Float16* wqT = xh  + (size_t)BSR * DD;                 // DD*DD
  _Float16* wkT = wqT + (size_t)DD * DD;
  _Float16* wvT = wkT + (size_t)DD * DD;
  _Float16* qh  = wvT + (size_t)DD * DD;                  // BSR*DD
  _Float16* kh  = qh  + (size_t)BSR * DD;                 // BSR*DD
  _Float16* vT  = kh  + (size_t)BSR * DD;                 // B*DD*SS
  _Float16* wh  = vT  + (size_t)BSR * DD;                 // B*SS*SS

  // 1) precision conversion / weight transpose
  cvt_x_kernel<<<(BSR * DD) / 256, 256, 0, stream>>>(x, xh);
  cvt_w_kernel<<<(DD * DD) / 256, 256, 0, stream>>>(Wq, Wk, Wv, wqT, wkT, wvT);

  // 2) QKV projections: 256*16 wave-tiles per matrix, 4 waves/block, y = which
  qkv_gemm_kernel<<<dim3(((BSR / 32) * (DD / 64)) / 4, 3), 128, 0, stream>>>(
      xh, wqT, wkT, wvT, qh, kh, vT);

  // 3) causal scores: 64*64 wave-tiles per batch (upper triangle early-outs)
  scores_kernel<<<dim3(((SS / 32) * (SS / 32)) / 4, BB), 128, 0, stream>>>(
      qh, kh, scoresOut);

  // 4) softmax over each of the B*S rows
  softmax_kernel<<<BB * SS, 256, 0, stream>>>(scoresOut, wh);

  // 5) att_output = weights @ v: 64*16 wave-tiles per batch
  av_gemm_kernel<<<dim3(((SS / 32) * (DD / 64)) / 4, BB), 128, 0, stream>>>(
      wh, vT, attOut);
}
